// CausalSelfAttention_8237747274097
// MI455X (gfx1250) — matrix operand order repaired and checked by reading, not compiled, not run on hardware
//
#include <hip/hip_runtime.h>
#include <hip/hip_bf16.h>

typedef __bf16 bf16;
typedef __attribute__((ext_vector_type(16))) __bf16 v16bf;
typedef __attribute__((ext_vector_type(8)))  float  v8f;
typedef __attribute__((__vector_size__(16))) int    i32x4v;   // matches builtin param

#define T_SEQ  2048
#define NHEAD  16
#define DMODEL 1024
#define HDIM   64

#define AS1 __attribute__((address_space(1)))
#define AS3 __attribute__((address_space(3)))

#if defined(__gfx1250__) && __has_builtin(__builtin_amdgcn_global_load_async_to_lds_b128)
#define HAVE_ASYNC_COPY 1
#else
#define HAVE_ASYNC_COPY 0
#endif

// 16-byte global -> LDS copy: async (ASYNCcnt) when available, else via VGPRs.
__device__ __forceinline__ void copy16_g2l(const bf16* gsrc, bf16* ldst) {
#if HAVE_ASYNC_COPY
  __builtin_amdgcn_global_load_async_to_lds_b128(
      (AS1 i32x4v*)gsrc, (AS3 i32x4v*)ldst, 0, 0);
#else
  *(uint4*)ldst = *(const uint4*)gsrc;
#endif
}

__device__ __forceinline__ void async_fence() {
#if HAVE_ASYNC_COPY
#if __has_builtin(__builtin_amdgcn_s_wait_asynccnt)
  __builtin_amdgcn_s_wait_asynccnt(0);
#else
  asm volatile("s_wait_asynccnt 0" ::: "memory");
#endif
#endif
}

// ---------------------------------------------------------------------------
// Fragment gathers from LDS, per CDNA5 WMMA VGPR layouts (CDNA5 ISA):
// A (16x32, MxK): lanes 0-15 M=lane, K {0..7,16..23}; lanes 16-31 M=lane-16,
//   K {8..15,24..31}; VGPR i holds one K pair.
// B (32x16, KxN): lanes 0-15 N=lane K {0..7,16..23}; lanes 16-31 N=lane-16 K {8..15,24..31};
//   read from an LDS buffer stored [N][K].
// C/D (16x16 f32): N = lane&15, M = vgpr + 8*(lane>=16).
// ---------------------------------------------------------------------------
__device__ __forceinline__ v16bf lds_load_a_frag(const bf16* base, int ld) {
  const int lane = threadIdx.x & 31;
  const int row  = lane & 15;
  const int hi8  = (lane >> 4) << 3;
  union { unsigned u[8]; v16bf v; } r;
#pragma unroll
  for (int i = 0; i < 8; ++i) {
    const int k = ((i >> 2) << 4) + hi8 + ((i & 3) << 1);
    r.u[i] = *(const unsigned*)(base + row * ld + k);
  }
  return r.v;
}

__device__ __forceinline__ v16bf lds_load_b_frag(const bf16* baseNT, int ld) {
  const int lane = threadIdx.x & 31;
  const int n    = lane & 15;
  const int hk   = (lane >> 4) << 3;
  union { unsigned u[8]; v16bf v; } r;
#pragma unroll
  for (int i = 0; i < 8; ++i) {
    const int k = ((i >> 2) << 4) + hk + ((i & 3) << 1);
    r.u[i] = *(const unsigned*)(baseNT + n * ld + k);
  }
  return r.v;
}

// ---------------------------------------------------------------------------
// One-shot precision/layout conversion kernels (memory-bound, run once).
// ---------------------------------------------------------------------------
__global__ __launch_bounds__(256)
void convert_f32_bf16(const float* __restrict__ in, bf16* __restrict__ out,
                      int n) {
  const int i = blockIdx.x * 256 + threadIdx.x;
  if (i < n) out[i] = (bf16)in[i];
}

// in: [K][N] f32 row-major  ->  out: [N][K] bf16  (N compile-time: div folds)
template <int N>
__global__ __launch_bounds__(256)
void transpose_convert(const float* __restrict__ in, bf16* __restrict__ out,
                       int K) {
  const int i = blockIdx.x * 256 + threadIdx.x;
  if (i < K * N) {
    const int k = i / N, n = i - k * N;
    out[(size_t)n * K + k] = (bf16)in[i];
  }
}

// ---------------------------------------------------------------------------
// Kernel 1: qkv = x @ W_attn + b_attn, scattered to q/k/v [B,H,T,64] bf16.
// Block 128 threads (4 waves); block tile 128(M) x 64(N); each wave 32x64.
// A = xb [M][K] bf16, B = Wat [N][K] bf16 (pre-transposed).
// ---------------------------------------------------------------------------
__global__ __launch_bounds__(128)
void qkv_gemm_kernel(const bf16* __restrict__ xb, const bf16* __restrict__ Wat,
                     const float* __restrict__ ba,
                     bf16* __restrict__ qo, bf16* __restrict__ ko,
                     bf16* __restrict__ vo) {
  __shared__ __align__(16) bf16 As[128][32];
  __shared__ __align__(16) bf16 Bt[64][32];
  const int tid  = threadIdx.x;
  const int mb0  = blockIdx.y * 128;
  const int nb0  = blockIdx.x * 64;
  const int wave = tid >> 5;
  const int lane = tid & 31;
  v8f acc[2][4] = {};

  for (int k0 = 0; k0 < DMODEL; k0 += 32) {
    __syncthreads();
    // A tile: 128 rows x 64B = 512 x 16B units; constant trip count (4/thread)
#pragma unroll
    for (int it = 0; it < 4; ++it) {
      const int u = tid + it * 128;
      const int r = u >> 2, seg = (u & 3) << 3;
      copy16_g2l(xb + (size_t)(mb0 + r) * DMODEL + k0 + seg, &As[r][seg]);
    }
    // B tile: 64 rows x 64B = 256 x 16B units (2/thread)
#pragma unroll
    for (int it = 0; it < 2; ++it) {
      const int u = tid + it * 128;
      const int n = u >> 2, seg = (u & 3) << 3;
      copy16_g2l(Wat + (size_t)(nb0 + n) * DMODEL + k0 + seg, &Bt[n][seg]);
    }
    async_fence();
    __syncthreads();

    v16bf a0 = lds_load_a_frag(&As[wave * 32][0], 32);
    v16bf a1 = lds_load_a_frag(&As[wave * 32 + 16][0], 32);
#pragma unroll
    for (int t = 0; t < 4; ++t) {
      const v16bf b = lds_load_b_frag(&Bt[t * 16][0], 32);
      acc[0][t] = __builtin_amdgcn_wmma_f32_16x16x32_bf16(
          false, a0, false, b, (short)0, acc[0][t], false, false);
      acc[1][t] = __builtin_amdgcn_wmma_f32_16x16x32_bf16(
          false, a1, false, b, (short)0, acc[1][t], false, false);
    }
  }

  const int colLo = lane & 15;
  const int rowHi = (lane >> 4) << 3;
#pragma unroll
  for (int su = 0; su < 2; ++su)
#pragma unroll
    for (int t = 0; t < 4; ++t)
#pragma unroll
      for (int i = 0; i < 8; ++i) {
        const int m = mb0 + wave * 32 + su * 16 + rowHi + i;
        const int n = nb0 + t * 16 + colLo;
        const float val = acc[su][t][i] + ba[n];
        const int sel = n >> 10, nn = n & 1023;
        const int head = nn >> 6, dim = nn & 63;
        const int bb = m >> 11, tt = m & 2047;       // T = 2048
        const size_t dst =
            (((size_t)bb * NHEAD + head) * T_SEQ + tt) * HDIM + dim;
        if (sel == 0)      qo[dst] = (bf16)(val * 0.125f);   // fold 1/sqrt(64)
        else if (sel == 1) ko[dst] = (bf16)val;
        else               vo[dst] = (bf16)val;
      }
}

// ---------------------------------------------------------------------------
// Kernel 2: causal flash attention.  Block = 128 threads (4 waves), one block
// per (b, h, 64-row query tile); each wave owns 16 query rows.
// ---------------------------------------------------------------------------
__global__ __launch_bounds__(128)
void flash_attn_kernel(const bf16* __restrict__ q, const bf16* __restrict__ k,
                       const bf16* __restrict__ v, bf16* __restrict__ y) {
  __shared__ __align__(16) bf16 Qs[64][HDIM];
  __shared__ __align__(16) bf16 Ks[64][HDIM];   // [key_row][dim]
  __shared__ __align__(16) bf16 Vt[HDIM][64];   // [dim][key_row]
  __shared__ __align__(16) bf16 Ps[4][16][64];  // per-wave P tile

  const int tid  = threadIdx.x;
  const int wave = tid >> 5;
  const int lane = tid & 31;
  const int qb   = blockIdx.x * 64;
  const int h    = blockIdx.y;
  const int b    = blockIdx.z;
  const size_t headBase = ((size_t)b * NHEAD + h) * T_SEQ * HDIM;

  // stage Q: 64 rows x 128B = 512 x 16B units (4/thread)
#pragma unroll
  for (int it = 0; it < 4; ++it) {
    const int u = tid + it * 128;
    const int r = u >> 3, seg = (u & 7) << 3;
    copy16_g2l(q + headBase + (size_t)(qb + r) * HDIM + seg, &Qs[r][seg]);
  }
  async_fence();
  __syncthreads();
  v16bf qfrag[2];
  qfrag[0] = lds_load_a_frag(&Qs[wave * 16][0], HDIM);
  qfrag[1] = lds_load_a_frag(&Qs[wave * 16][32], HDIM);

  v8f oacc[4] = {};
  float mrow[8], lrow[8];
#pragma unroll
  for (int i = 0; i < 8; ++i) { mrow[i] = -__builtin_inff(); lrow[i] = 0.f; }

  const int colLo = lane & 15;
  const int rowHi = (lane >> 4) << 3;
  const int qrow0 = qb + wave * 16;

  for (int jt = 0; jt <= (int)blockIdx.x; ++jt) {
    const int j0 = jt * 64;
    __syncthreads();
    // K tile: raw async copy, natural layout
#pragma unroll
    for (int it = 0; it < 4; ++it) {
      const int u = tid + it * 128;
      const int r = u >> 3, seg = (u & 7) << 3;
      copy16_g2l(k + headBase + (size_t)(j0 + r) * HDIM + seg, &Ks[r][seg]);
    }
    // V tile: 16B vector loads, transposed scatter into Vt
#pragma unroll
    for (int it = 0; it < 4; ++it) {
      const int u = tid + it * 128;
      const int r = u >> 3, seg = (u & 7) << 3;
      union { uint4 q4; bf16 hv[8]; } w;
      w.q4 = *(const uint4*)(v + headBase + (size_t)(j0 + r) * HDIM + seg);
#pragma unroll
      for (int e = 0; e < 8; ++e) Vt[seg + e][r] = w.hv[e];
    }
    async_fence();
    __syncthreads();

    // S = Q K^T  (scale pre-folded into Q)
    v8f s[4];
#pragma unroll
    for (int t = 0; t < 4; ++t) {
      const v16bf b0 = lds_load_b_frag(&Ks[t * 16][0], HDIM);
      const v16bf b1 = lds_load_b_frag(&Ks[t * 16][32], HDIM);
      v8f z = {};
      z = __builtin_amdgcn_wmma_f32_16x16x32_bf16(
          false, qfrag[0], false, b0, (short)0, z, false, false);
      z = __builtin_amdgcn_wmma_f32_16x16x32_bf16(
          false, qfrag[1], false, b1, (short)0, z, false, false);
      s[t] = z;
    }

    if (jt == (int)blockIdx.x) {     // causal mask only on the diagonal block
#pragma unroll
      for (int t = 0; t < 4; ++t)
#pragma unroll
        for (int i = 0; i < 8; ++i) {
          const int qi = qrow0 + rowHi + i;
          const int kj = j0 + t * 16 + colLo;
          if (kj > qi) s[t][i] = -__builtin_inff();
        }
    }

    // online softmax
    float mnew[8], alpha[8], rs[8];
#pragma unroll
    for (int i = 0; i < 8; ++i) {
      float rm = fmaxf(fmaxf(s[0][i], s[1][i]), fmaxf(s[2][i], s[3][i]));
#pragma unroll
      for (int off = 1; off < 16; off <<= 1)
        rm = fmaxf(rm, __shfl_xor(rm, off, 32));
      mnew[i]  = fmaxf(mrow[i], rm);
      alpha[i] = __expf(mrow[i] - mnew[i]);
      rs[i]    = 0.f;
    }
#pragma unroll
    for (int t = 0; t < 4; ++t)
#pragma unroll
      for (int i = 0; i < 8; ++i) {
        const float p = __expf(s[t][i] - mnew[i]);
        rs[i] += p;
        Ps[wave][rowHi + i][t * 16 + colLo] = (bf16)p;
      }
#pragma unroll
    for (int i = 0; i < 8; ++i) {
#pragma unroll
      for (int off = 1; off < 16; off <<= 1)
        rs[i] += __shfl_xor(rs[i], off, 32);
      lrow[i] = lrow[i] * alpha[i] + rs[i];
      mrow[i] = mnew[i];
    }
    __asm__ volatile("" ::: "memory");   // order P stores before reloads

    // O = O*alpha + P V
    const v16bf pf0 = lds_load_a_frag(&Ps[wave][0][0], 64);
    const v16bf pf1 = lds_load_a_frag(&Ps[wave][0][32], 64);
#pragma unroll
    for (int t = 0; t < 4; ++t) {
#pragma unroll
      for (int i = 0; i < 8; ++i) oacc[t][i] *= alpha[i];
      const v16bf vb0 = lds_load_b_frag(&Vt[t * 16][0], 64);
      const v16bf vb1 = lds_load_b_frag(&Vt[t * 16][32], 64);
      oacc[t] = __builtin_amdgcn_wmma_f32_16x16x32_bf16(
          false, pf0, false, vb0, (short)0, oacc[t], false, false);
      oacc[t] = __builtin_amdgcn_wmma_f32_16x16x32_bf16(
          false, pf1, false, vb1, (short)0, oacc[t], false, false);
    }
  }

  // y[b, t, h*64 + dim] = O / l
#pragma unroll
  for (int t = 0; t < 4; ++t)
#pragma unroll
    for (int i = 0; i < 8; ++i) {
      const int m = qrow0 + rowHi + i;
      const float val = oacc[t][i] / lrow[i];
      const size_t idx =
          ((size_t)b * T_SEQ + m) * DMODEL + h * HDIM + t * 16 + colLo;
      y[idx] = (bf16)val;
    }
}

// ---------------------------------------------------------------------------
// Kernel 3: out = y @ W_proj + b_proj  (bf16 x bf16 -> f32)
// Same tiling as kernel 1; B = Wpt [N][K] bf16 (pre-transposed).
// ---------------------------------------------------------------------------
__global__ __launch_bounds__(128)
void proj_gemm_kernel(const bf16* __restrict__ y, const bf16* __restrict__ Wpt,
                      const float* __restrict__ bp, float* __restrict__ out) {
  __shared__ __align__(16) bf16 As[128][32];
  __shared__ __align__(16) bf16 Bt[64][32];
  const int tid  = threadIdx.x;
  const int mb0  = blockIdx.y * 128;
  const int nb0  = blockIdx.x * 64;
  const int wave = tid >> 5;
  const int lane = tid & 31;
  v8f acc[2][4] = {};

  for (int k0 = 0; k0 < DMODEL; k0 += 32) {
    __syncthreads();
#pragma unroll
    for (int it = 0; it < 4; ++it) {
      const int u = tid + it * 128;
      const int r = u >> 2, seg = (u & 3) << 3;
      copy16_g2l(y + (size_t)(mb0 + r) * DMODEL + k0 + seg, &As[r][seg]);
    }
#pragma unroll
    for (int it = 0; it < 2; ++it) {
      const int u = tid + it * 128;
      const int n = u >> 2, seg = (u & 3) << 3;
      copy16_g2l(Wpt + (size_t)(nb0 + n) * DMODEL + k0 + seg, &Bt[n][seg]);
    }
    async_fence();
    __syncthreads();

    v16bf a0 = lds_load_a_frag(&As[wave * 32][0], 32);
    v16bf a1 = lds_load_a_frag(&As[wave * 32 + 16][0], 32);
#pragma unroll
    for (int t = 0; t < 4; ++t) {
      const v16bf b = lds_load_b_frag(&Bt[t * 16][0], 32);
      acc[0][t] = __builtin_amdgcn_wmma_f32_16x16x32_bf16(
          false, a0, false, b, (short)0, acc[0][t], false, false);
      acc[1][t] = __builtin_amdgcn_wmma_f32_16x16x32_bf16(
          false, a1, false, b, (short)0, acc[1][t], false, false);
    }
  }

  const int colLo = lane & 15;
  const int rowHi = (lane >> 4) << 3;
#pragma unroll
  for (int su = 0; su < 2; ++su)
#pragma unroll
    for (int t = 0; t < 4; ++t)
#pragma unroll
      for (int i = 0; i < 8; ++i) {
        const int m = mb0 + wave * 32 + su * 16 + rowHi + i;
        const int n = nb0 + t * 16 + colLo;
        out[(size_t)m * DMODEL + n] = acc[su][t][i] + bp[n];
      }
}

// ---------------------------------------------------------------------------
extern "C" void kernel_launch(void* const* d_in, const int* in_sizes, int n_in,
                              void* d_out, int out_size, void* d_ws, size_t ws_size,
                              hipStream_t stream) {
  const float* x  = (const float*)d_in[0];
  const float* Wa = (const float*)d_in[1];
  const float* ba = (const float*)d_in[2];
  const float* Wp = (const float*)d_in[3];
  const float* bp = (const float*)d_in[4];
  float* out = (float*)d_out;

  const size_t M      = (size_t)4 * T_SEQ;          // 8192 tokens
  const size_t X_E    = M * DMODEL;                 // 8M
  const size_t WA_E   = (size_t)3 * DMODEL * DMODEL;// 3M
  const size_t WP_E   = (size_t)DMODEL * DMODEL;    // 1M
  bf16* xb  = (bf16*)d_ws;            // [M][K]     (later reused as y)
  bf16* Wat = xb  + X_E;              // [3D][K]    transposed
  bf16* Wpt = Wat + WA_E;             // [D][K]     transposed
  bf16* q   = Wpt + WP_E;             // [B,H,T,64]
  bf16* k   = q   + X_E;
  bf16* v   = k   + X_E;
  bf16* y   = xb;                     // alias: xb dead after qkv GEMM

  // one-shot conversions / layout changes
  convert_f32_bf16<<<dim3((unsigned)((X_E + 255) / 256)), 256, 0, stream>>>(
      x, xb, (int)X_E);
  transpose_convert<3 * DMODEL>
      <<<dim3((unsigned)((WA_E + 255) / 256)), 256, 0, stream>>>(Wa, Wat,
                                                                 DMODEL);
  transpose_convert<DMODEL>
      <<<dim3((unsigned)((WP_E + 255) / 256)), 256, 0, stream>>>(Wp, Wpt,
                                                                 DMODEL);

  // qkv: M=8192, N=3072, K=1024
  qkv_gemm_kernel<<<dim3(48, 64), 128, 0, stream>>>(xb, Wat, ba, q, k, v);
  // attention: one block per (qtile, head, batch)
  flash_attn_kernel<<<dim3(T_SEQ / 64, NHEAD, 4), 128, 0, stream>>>(q, k, v, y);
  // proj: M=8192, N=1024, K=1024
  proj_gemm_kernel<<<dim3(16, 64), 128, 0, stream>>>(y, Wpt, bp, out);
}
